// ResAttnBlock_22316650070627
// MI455X (gfx1250) — matrix operand order repaired and checked by reading, not compiled, not run on hardware
//
#include <hip/hip_runtime.h>
#include <hip/hip_bf16.h>

typedef __attribute__((ext_vector_type(16))) _Float16 v16h;
typedef __attribute__((ext_vector_type(8)))  float    v8f;

#define WMMA16(a, b, c) \
  __builtin_amdgcn_wmma_f32_16x16x32_f16(false, (a), false, (b), (short)0, (c), false, false)

static constexpr int B_   = 32;
static constexpr int DIM  = 256;
static constexpr int HW   = 32;
static constexpr int NTOK = HW * HW;            // 1024
static constexpr int ROWS = B_ * NTOK;          // 32768 token rows
static constexpr float EPS   = 1e-5f;
static constexpr float SLOPE = 0.2f;

// ---------------------------------------------------------------------------
// WMMA fragment helpers (wave32, 16x16x32 f16 -> f32). A/B k order: contiguous k = 16g + j per lane half g.
// ---------------------------------------------------------------------------
__device__ inline v8f zero8() {
  v8f z;
#pragma unroll
  for (int i = 0; i < 8; ++i) z[i] = 0.0f;
  return z;
}

// A matrix 16x32 f16, row-major source with row stride lda (halves).
// Per lane: one contiguous 16-half run (k = 16g..16g+15) -> vectorizes to 2x b128 loads.
__device__ inline v16h load_a_frag(const _Float16* a, int lda) {
  const int lane = threadIdx.x & 31;
  const int m = lane & 15, g = lane >> 4;
  v16h r;
#pragma unroll
  for (int j = 0; j < 16; ++j) {
    const int k = 16 * g + j;
    r[j] = a[m * lda + k];
  }
  return r;
}

// B matrix fragment from LDS pre-swizzled to fragment order: src[lane*16 + j].
// One contiguous 32-byte run per lane -> 2x ds_load_b128.
__device__ inline v16h load_b_frag_sw(const _Float16* b) {
  const int lane = threadIdx.x & 31;
  v16h r;
#pragma unroll
  for (int j = 0; j < 16; ++j) r[j] = b[lane * 16 + j];
  return r;
}

// B fragment from row-major-[n][k] source (e.g. K tiles): element (k,n) at
// src[n*ldb + k]; slots k=16g..16g+15 contiguous per lane -> 2x ds_load_b128.
__device__ inline v16h load_bt_frag(const _Float16* b, int ldb) {
  const int lane = threadIdx.x & 31;
  const int n = lane & 15, g = lane >> 4;
  v16h r;
#pragma unroll
  for (int j = 0; j < 16; ++j) r[j] = b[n * ldb + 16 * g + j];
  return r;
}

// swizzled-store position for matrix element (k, col%128) inside a 32xK-chunk
// B tile region laid out [tile][lane][slot]
__device__ inline int bsw_index(int kk, int ncol) {
  const int nt = ncol >> 4, n = ncol & 15;
  const int g2 = kk >> 4, j2 = kk & 15;
  return ((nt * 32) + n + 16 * g2) * 16 + j2;
}

// ---------------------------------------------------------------------------
// K0: zero BN statistic accumulators (4 x 256 floats)
// ---------------------------------------------------------------------------
__global__ void init_stats(float* stats) {
  stats[threadIdx.x] = 0.0f;
}

// ---------------------------------------------------------------------------
// K1: conv(k2,s2) as im2col GEMM. Block tile: 128 rows x 128 cols, 8 waves,
// each wave 16 rows x 128 cols (8 N-tiles -> 64 acc VGPRs, no spills).
// grid = (256, 2)
// ---------------------------------------------------------------------------
__global__ void __launch_bounds__(256, 1)
conv_gemm(const float* __restrict__ c, const float* __restrict__ w,
          const float* __restrict__ bias, float* __restrict__ c1a,
          float* __restrict__ bn1s, float* __restrict__ bn1q) {
  __shared__ _Float16 As[128][40];        // 128 rows x 32 k (+pad)
  __shared__ _Float16 Bsw[8 * 32 * 16];   // 8 N-tiles, fragment-order
  const int tid = threadIdx.x, lane = tid & 31, wv = tid >> 5;
  const int g = lane >> 4;
  const int Rbase = blockIdx.x * 128;
  const int Cbase = blockIdx.y * 128;

  v8f acc[8];
#pragma unroll
  for (int i = 0; i < 8; ++i) acc[i] = zero8();

  for (int ch = 0; ch < 8; ++ch) {
    __syncthreads();
    // A tile: im2col of c. row R=(b,h,w); col kcol = ic*4 + kh*2 + kw
#pragma unroll
    for (int i = 0; i < 16; ++i) {
      const int idx = i * 256 + tid;
      const int r = idx & 127, kk = idx >> 7;
      const int R = Rbase + r;
      const int bb = R >> 10, n = R & 1023;
      const int hh = n >> 5, ww = n & 31;
      const int kcol = ch * 32 + kk;
      const int ic = kcol >> 2, kh = (kcol >> 1) & 1, kw = kcol & 1;
      As[r][kk] = (_Float16)c[(size_t)bb * 262144 + (size_t)ic * 4096 +
                              (2 * hh + kh) * 64 + (2 * ww + kw)];
    }
    // B tile (swizzled): B[k][oc] = w[oc*256 + k]
#pragma unroll
    for (int i = 0; i < 16; ++i) {
      const int idx = i * 256 + tid;
      const int kk = idx & 31, ncol = idx >> 5;  // ncol in [0,128)
      Bsw[bsw_index(kk, ncol)] =
          (_Float16)w[(Cbase + ncol) * 256 + ch * 32 + kk];
    }
    __syncthreads();

    const v16h a = load_a_frag(&As[wv * 16][0], 40);
#pragma unroll
    for (int nt = 0; nt < 8; ++nt) {
      const v16h bf = load_b_frag_sw(&Bsw[nt * 512]);
      acc[nt] = WMMA16(a, bf, acc[nt]);
    }
  }

  // epilogue: +bias, LeakyReLU, store, BN1 stats
#pragma unroll
  for (int nt = 0; nt < 8; ++nt) {
    const int col = Cbase + nt * 16 + (lane & 15);
    const float bv = bias[col];
    float cs = 0.0f, cq = 0.0f;
#pragma unroll
    for (int r = 0; r < 8; ++r) {
      float val = acc[nt][r] + bv;
      val = (val >= 0.0f) ? val : SLOPE * val;
      c1a[(size_t)(Rbase + wv * 16 + r + 8 * g) * DIM + col] = val;
      cs += val;
      cq += val * val;
    }
    cs += __shfl_xor(cs, 16, 32);
    cq += __shfl_xor(cq, 16, 32);
    if (lane < 16) {
      atomicAdd(&bn1s[col], cs);
      atomicAdd(&bn1q[col], cq);
    }
  }
}

// ---------------------------------------------------------------------------
// K2: K/V projection GEMM: out[b,n,e] = sum_d x[b,d,n] * W[e,d] + bias[e]
// Same tiling as conv_gemm. grid = (256, 2)
// ---------------------------------------------------------------------------
__global__ void __launch_bounds__(256, 1)
proj_gemm(const float* __restrict__ x, const float* __restrict__ w,
          const float* __restrict__ bias, _Float16* __restrict__ out) {
  __shared__ _Float16 As[128][40];
  __shared__ _Float16 Bsw[8 * 32 * 16];
  const int tid = threadIdx.x, lane = tid & 31, wv = tid >> 5;
  const int g = lane >> 4;
  const int Rbase = blockIdx.x * 128;
  const int Cbase = blockIdx.y * 128;

  v8f acc[8];
#pragma unroll
  for (int i = 0; i < 8; ++i) acc[i] = zero8();

  for (int ch = 0; ch < 8; ++ch) {
    __syncthreads();
#pragma unroll
    for (int i = 0; i < 16; ++i) {
      const int idx = i * 256 + tid;
      const int r = idx & 127, kk = idx >> 7;
      const int R = Rbase + r;
      const int bb = R >> 10, n = R & 1023;
      const int d = ch * 32 + kk;
      As[r][kk] = (_Float16)x[(size_t)bb * 262144 + (size_t)d * 1024 + n];
    }
#pragma unroll
    for (int i = 0; i < 16; ++i) {
      const int idx = i * 256 + tid;
      const int kk = idx & 31, ncol = idx >> 5;
      Bsw[bsw_index(kk, ncol)] =
          (_Float16)w[(Cbase + ncol) * 256 + ch * 32 + kk];
    }
    __syncthreads();

    const v16h a = load_a_frag(&As[wv * 16][0], 40);
#pragma unroll
    for (int nt = 0; nt < 8; ++nt) {
      const v16h bf = load_b_frag_sw(&Bsw[nt * 512]);
      acc[nt] = WMMA16(a, bf, acc[nt]);
    }
  }

#pragma unroll
  for (int nt = 0; nt < 8; ++nt) {
    const int col = Cbase + nt * 16 + (lane & 15);
    const float bv = bias[col];
#pragma unroll
    for (int r = 0; r < 8; ++r) {
      out[(size_t)(Rbase + wv * 16 + r + 8 * g) * DIM + col] =
          (_Float16)(acc[nt][r] + bv);
    }
  }
}

// ---------------------------------------------------------------------------
// K3: BN1 finalize -> Q in f16
// ---------------------------------------------------------------------------
__global__ void bn1_apply(const float* __restrict__ c1a, const float* __restrict__ s,
                          const float* __restrict__ q, const float* __restrict__ gam,
                          const float* __restrict__ bet, _Float16* __restrict__ qb) {
  const int gid = blockIdx.x * 256 + threadIdx.x;
  const int d = gid & 255;
  const float cnt = (float)ROWS;
  const float m = s[d] / cnt;
  const float v = q[d] / cnt - m * m;
  const float val = (c1a[gid] - m) * rsqrtf(v + EPS) * gam[d] + bet[d];
  qb[gid] = (_Float16)val;
}

// ---------------------------------------------------------------------------
// K4: flash attention + residual + LeakyReLU + BN2 stats
// grid = B * 8 (128-row tiles); block = 256 threads = 8 waves x 16 rows.
// K tile copied global->LDS with CDNA5 async-DMA (ASYNCcnt); V tile swizzled
// to fragment order on the VALU path. Q A-fragments re-loaded from global
// (L2-resident) per chunk to keep registers at acc(128) + softmax state(16).
// ---------------------------------------------------------------------------
__global__ void __launch_bounds__(256, 1)
attn_kernel(const _Float16* __restrict__ qbuf, const _Float16* __restrict__ kbuf,
            const _Float16* __restrict__ vbuf, const float* __restrict__ x,
            float* __restrict__ ybuf, float* __restrict__ bn2s,
            float* __restrict__ bn2q) {
  __shared__ _Float16 Ks[32][264];         // natural layout (bt_frag reads)
  __shared__ _Float16 Vsw[16 * 32 * 16];   // fragment-order, 16 N-tiles
  __shared__ _Float16 Ps[8][16][32];       // per-wave P tile
  const int tid = threadIdx.x, lane = tid & 31, wv = tid >> 5;
  const int m = lane & 15, g = lane >> 4;
  const int b = blockIdx.x >> 3;
  const int rowbase = (blockIdx.x & 7) * 128 + wv * 16;  // token row in batch

  const _Float16* qrow = qbuf + ((size_t)b * NTOK + rowbase) * DIM;

  v8f acc[16];
#pragma unroll
  for (int i = 0; i < 16; ++i) acc[i] = zero8();
  float rmax[8], rsum[8];
#pragma unroll
  for (int r = 0; r < 8; ++r) { rmax[r] = -3.0e38f; rsum[r] = 0.0f; }

  for (int kb = 0; kb < 32; ++kb) {
    __syncthreads();
    const _Float16* ksrc = kbuf + ((size_t)b * NTOK + kb * 32) * DIM;
    const _Float16* vsrc = vbuf + ((size_t)b * NTOK + kb * 32) * DIM;

    // --- K tile: async global->LDS copy (16B per lane per op, ASYNCcnt) ---
#pragma unroll
    for (int i = 0; i < 4; ++i) {
      const int cidx = i * 256 + tid;        // 1024 x 16-byte chunks
      const int krow = cidx >> 5;            // 32 chunks per 512B row
      const int dof  = (cidx & 31) * 8;      // halves within row
      const unsigned int ldsa = (unsigned int)(size_t)&Ks[krow][dof];
      const _Float16* gsa = ksrc + krow * 256 + dof;
      asm volatile("global_load_async_to_lds_b128 %0, %1, off"
                   :: "v"(ldsa), "v"(gsa) : "memory");
    }

    // --- V tile: fragment-order swizzle (element gather) ---
#pragma unroll
    for (int i = 0; i < 32; ++i) {
      const int idx = i * 256 + tid;
      const int k2 = idx >> 8, d2 = idx & 255;
      Vsw[bsw_index(k2, d2 & 127) + (d2 >> 7) * 4096] = vsrc[idx];
    }

    // async copies are not covered by the workgroup barrier's DS waits
    asm volatile("s_wait_asynccnt 0x0" ::: "memory");
    __syncthreads();

    // S = Q K^T for two 16-key tiles
    v8f s0 = zero8(), s1 = zero8();
#pragma unroll
    for (int ch = 0; ch < 8; ++ch) {
      const v16h a = load_a_frag(qrow + ch * 32, DIM);
      const v16h bk0 = load_bt_frag(&Ks[0][ch * 32], 264);
      const v16h bk1 = load_bt_frag(&Ks[16][ch * 32], 264);
      s0 = WMMA16(a, bk0, s0);
      s1 = WMMA16(a, bk1, s1);
    }

    // online softmax update
#pragma unroll
    for (int r = 0; r < 8; ++r) {
      float t = fmaxf(s0[r], s1[r]);
      t = fmaxf(t, __shfl_xor(t, 1, 32));
      t = fmaxf(t, __shfl_xor(t, 2, 32));
      t = fmaxf(t, __shfl_xor(t, 4, 32));
      t = fmaxf(t, __shfl_xor(t, 8, 32));
      const float nmax = fmaxf(rmax[r], t);
      const float sc = __expf(rmax[r] - nmax);
      const float p0 = __expf(s0[r] - nmax);
      const float p1 = __expf(s1[r] - nmax);
      float ps = p0 + p1;
      ps += __shfl_xor(ps, 1, 32);
      ps += __shfl_xor(ps, 2, 32);
      ps += __shfl_xor(ps, 4, 32);
      ps += __shfl_xor(ps, 8, 32);
      rsum[r] = rsum[r] * sc + ps;
      rmax[r] = nmax;
#pragma unroll
      for (int nt = 0; nt < 16; ++nt) acc[nt][r] *= sc;
      Ps[wv][r + 8 * g][m]      = (_Float16)p0;
      Ps[wv][r + 8 * g][m + 16] = (_Float16)p1;
    }

    // O += P V
    const v16h pa = load_a_frag(&Ps[wv][0][0], 32);
#pragma unroll
    for (int nt = 0; nt < 16; ++nt) {
      const v16h bv = load_b_frag_sw(&Vsw[nt * 512]);
      acc[nt] = WMMA16(pa, bv, acc[nt]);
    }
  }

  // epilogue: /rowsum, +x residual, LeakyReLU, store + BN2 stats
  float inv[8];
#pragma unroll
  for (int r = 0; r < 8; ++r) inv[r] = 1.0f / rsum[r];
  const float* xb = x + (size_t)b * 262144;
#pragma unroll
  for (int nt = 0; nt < 16; ++nt) {
    const int d = nt * 16 + m;
    float cs = 0.0f, cq = 0.0f;
#pragma unroll
    for (int r = 0; r < 8; ++r) {
      const int n = rowbase + r + 8 * g;
      float val = acc[nt][r] * inv[r] + xb[(size_t)d * 1024 + n];
      val = (val >= 0.0f) ? val : SLOPE * val;
      ybuf[((size_t)b * NTOK + n) * DIM + d] = val;
      cs += val;
      cq += val * val;
    }
    cs += __shfl_xor(cs, 16, 32);
    cq += __shfl_xor(cq, 16, 32);
    if (lane < 16) {
      atomicAdd(&bn2s[d], cs);
      atomicAdd(&bn2q[d], cq);
    }
  }
}

// ---------------------------------------------------------------------------
// K5: BN2 finalize, transpose [b,n,d] -> NCHW output
// ---------------------------------------------------------------------------
__global__ void bn2_apply(const float* __restrict__ ybuf, const float* __restrict__ s,
                          const float* __restrict__ q, const float* __restrict__ gam,
                          const float* __restrict__ bet, float* __restrict__ out) {
  const int gid = blockIdx.x * 256 + threadIdx.x;   // NCHW linear index
  const int b = gid >> 18;
  const int d = (gid >> 10) & 255;
  const int n = gid & 1023;
  const float cnt = (float)ROWS;
  const float m = s[d] / cnt;
  const float v = q[d] / cnt - m * m;
  const float y = ybuf[(size_t)b * 262144 + (size_t)n * DIM + d];
  out[gid] = (y - m) * rsqrtf(v + EPS) * gam[d] + bet[d];
}

// ---------------------------------------------------------------------------
extern "C" void kernel_launch(void* const* d_in, const int* in_sizes, int n_in,
                              void* d_out, int out_size, void* d_ws, size_t ws_size,
                              hipStream_t stream) {
  const float* x      = (const float*)d_in[0];
  const float* c      = (const float*)d_in[1];
  const float* conv_w = (const float*)d_in[2];
  const float* conv_b = (const float*)d_in[3];
  const float* bn1_g  = (const float*)d_in[4];
  const float* bn1_b  = (const float*)d_in[5];
  const float* key_w  = (const float*)d_in[6];
  const float* key_b  = (const float*)d_in[7];
  const float* val_w  = (const float*)d_in[8];
  const float* val_b  = (const float*)d_in[9];
  const float* bn2_g  = (const float*)d_in[10];
  const float* bn2_b  = (const float*)d_in[11];

  char* ws = (char*)d_ws;
  float*    c1a   = (float*)ws;                              // 32 MiB
  float*    ybuf  = (float*)(ws + 33554432);                 // 32 MiB
  _Float16* qb    = (_Float16*)(ws + 67108864);              // 16 MiB
  _Float16* kb    = (_Float16*)(ws + 83886080);              // 16 MiB
  _Float16* vb    = (_Float16*)(ws + 100663296);             // 16 MiB
  float*    stats = (float*)(ws + 117440512);                // 4 x 256 floats
  float* bn1s = stats, *bn1q = stats + 256;
  float* bn2s = stats + 512, *bn2q = stats + 768;

  init_stats<<<1, 1024, 0, stream>>>(stats);
  conv_gemm<<<dim3(256, 2), 256, 0, stream>>>(c, conv_w, conv_b, c1a, bn1s, bn1q);
  proj_gemm<<<dim3(256, 2), 256, 0, stream>>>(x, key_w, key_b, kb);
  proj_gemm<<<dim3(256, 2), 256, 0, stream>>>(x, val_w, val_b, vb);
  bn1_apply<<<32768, 256, 0, stream>>>(c1a, bn1s, bn1q, bn1_g, bn1_b, qb);
  attn_kernel<<<256, 256, 0, stream>>>(qb, kb, vb, x, ybuf, bn2s, bn2q);
  bn2_apply<<<32768, 256, 0, stream>>>(ybuf, bn2s, bn2q, bn2_g, bn2_b, (float*)d_out);
}
